// MultiHeadAttention_27668179321185
// MI455X (gfx1250) — matrix operand order repaired and checked by reading, not compiled, not run on hardware
//
#include <hip/hip_runtime.h>
#include <hip/hip_bf16.h>

typedef __attribute__((ext_vector_type(16))) _Float16 v16h;
typedef __attribute__((ext_vector_type(8)))  float    v8f;

#define D_MODEL 1024
#define NUM_HEADS 16
#define D_K 64
#define SEQ 2048
#define BATCH 2
#define ROWS (BATCH*SEQ)     /* 4096 */
#define NQKV (3*D_MODEL)     /* 3072 */
#define CHUNKS (SEQ/32)      /* 64 key chunks */

// ---------- WMMA fragment loaders (16x16x32 f16, this file's contiguous k order for A and B, wave32) ----------
// A (16x32 f16): lane L holds row M=L%16; elements 0..15 -> K=(L/16)*16+0..15.
__device__ inline v16h load_a_rowmajor(const _Float16* p, int ld, int row, int k0, int hf) {
  const _Float16* r = p + (size_t)row * ld + k0;
  v16h a;
  ((uint4*)&a)[0] = *(const uint4*)(r + hf * 16);
  ((uint4*)&a)[1] = *(const uint4*)(r + hf * 16 + 8);
  return a;
}
// B (32x16 f16) from a K-contiguous ("B^T") array bt[n][k]: lane L holds column
// N=L%16, K=(L/16)*16 + e for e=0..15 -> 32 contiguous bytes. Caller passes
// k0 already including (L/16)*16. Works for both global and LDS pointers.
__device__ inline v16h load_b_kcontig(const _Float16* bt, int ld, int n, int k0) {
  const _Float16* r = bt + (size_t)n * ld + k0;
  v16h b;
  ((uint4*)&b)[0] = ((const uint4*)r)[0];
  ((uint4*)&b)[1] = ((const uint4*)r)[1];
  return b;
}

// ---------- CDNA5 async global->LDS copy (16B per lane, ASYNCcnt-tracked) ----------
// LDS aperture: generic address low 32 bits == LDS byte address (ISA 10.2).
__device__ inline void async_copy16(const _Float16* gsrc, _Float16* ldst) {
  unsigned l = (unsigned)(uintptr_t)ldst;
  asm volatile("global_load_async_to_lds_b128 %0, %1, off"
               :: "v"(l), "v"(gsrc) : "memory");
}

// ---------------- LayerNorm (fp32 in, f16 normalized out) ----------------
__global__ void ln_kernel(const float* __restrict__ x, const float* __restrict__ g,
                          const float* __restrict__ beta, _Float16* __restrict__ xn) {
  __shared__ float red[256];
  int row = blockIdx.x, tid = threadIdx.x;
  const float* xr = x + (size_t)row * D_MODEL;
  float s = 0.f;
  for (int i = tid; i < D_MODEL; i += 256) s += xr[i];
  red[tid] = s; __syncthreads();
  for (int o = 128; o > 0; o >>= 1) { if (tid < o) red[tid] += red[tid + o]; __syncthreads(); }
  float mean = red[0] * (1.0f / D_MODEL); __syncthreads();
  float v = 0.f;
  for (int i = tid; i < D_MODEL; i += 256) { float d = xr[i] - mean; v += d * d; }
  red[tid] = v; __syncthreads();
  for (int o = 128; o > 0; o >>= 1) { if (tid < o) red[tid] += red[tid + o]; __syncthreads(); }
  float rstd = rsqrtf(red[0] * (1.0f / D_MODEL) + 1e-5f);
  _Float16* o = xn + (size_t)row * D_MODEL;
  for (int i = tid; i < D_MODEL; i += 256)
    o[i] = (_Float16)((xr[i] - mean) * rstd * g[i] + beta[i]);
}

// ---------------- Weight packing: wT[n][d] = W[d][n] as f16 ----------------
__global__ void pack_wqkv(const float* __restrict__ wq, const float* __restrict__ wk,
                          const float* __restrict__ wv, _Float16* __restrict__ wT) {
  int idx = blockIdx.x * 256 + threadIdx.x;
  if (idx >= NQKV * D_MODEL) return;
  int n = idx / D_MODEL, d = idx % D_MODEL;
  int mat = n >> 10, hk = n & 1023, h = hk >> 6, dk = hk & 63;
  const float* w = (mat == 0) ? wq : (mat == 1) ? wk : wv;   // (H, D, Dk)
  wT[idx] = (_Float16)w[((size_t)h * D_MODEL + d) * D_K + dk];
}
__global__ void pack_wo(const float* __restrict__ wo, _Float16* __restrict__ woT) {
  int idx = blockIdx.x * 256 + threadIdx.x;
  if (idx >= D_MODEL * D_MODEL) return;
  int n = idx / D_MODEL, w = idx % D_MODEL;                  // woT[n][w] = wo[w][n]
  woT[idx] = (_Float16)wo[(size_t)w * D_MODEL + n];
}

// ------- Fused QKV GEMM: (4096x1024) x (1024x3072), 32x64 tile / wave -------
__global__ __launch_bounds__(32) void qkv_kernel(
    const _Float16* __restrict__ xn, const _Float16* __restrict__ wT,
    const float* __restrict__ bq, const float* __restrict__ bk, const float* __restrict__ bv,
    _Float16* __restrict__ q, _Float16* __restrict__ kb, _Float16* __restrict__ vt) {
  int lane = threadIdx.x, lr = lane & 15, hf = lane >> 4;
  int m0 = blockIdx.y * 32, n0 = blockIdx.x * 64;
  v8f acc[2][4] = {};
  for (int kk = 0; kk < D_MODEL; kk += 32) {
    v16h a0 = load_a_rowmajor(xn, D_MODEL, m0 + lr,      kk, hf);
    v16h a1 = load_a_rowmajor(xn, D_MODEL, m0 + 16 + lr, kk, hf);
#pragma unroll
    for (int j = 0; j < 4; j++) {
      v16h b = load_b_kcontig(wT, D_MODEL, n0 + j * 16 + lr, kk + hf * 16);
      acc[0][j] = __builtin_amdgcn_wmma_f32_16x16x32_f16(false, a0, false, b, (short)0, acc[0][j], false, false);
      acc[1][j] = __builtin_amdgcn_wmma_f32_16x16x32_f16(false, a1, false, b, (short)0, acc[1][j], false, false);
    }
  }
#pragma unroll
  for (int j = 0; j < 4; j++) {
    int n = n0 + j * 16 + lr;          // C layout: N = lane%16
    int mat = n >> 10, idx = n & 1023, h = idx >> 6, dk = idx & 63;
#pragma unroll
    for (int i = 0; i < 2; i++) {
#pragma unroll
      for (int r = 0; r < 8; r++) {
        int m = m0 + i * 16 + r + 8 * hf;  // C layout: M = r + 8*(lane/16)
        int b_ = m >> 11, t = m & 2047;
        int bh = b_ * NUM_HEADS + h;
        float val = acc[i][j][r];
        if (mat == 0)
          q [((size_t)bh * SEQ + t) * D_K + dk] = (_Float16)((val + bq[idx]) * 0.125f); // fold 1/sqrt(64)
        else if (mat == 1)
          kb[((size_t)bh * SEQ + t) * D_K + dk] = (_Float16)(val + bk[idx]);
        else
          vt[((size_t)bh * D_K + dk) * SEQ + t] = (_Float16)(val + bv[idx]);            // V transposed
      }
    }
  }
}

// ---- Stage one 32-key chunk: K tile (32x64) + V^T tile (64x32), 4 async b128 / wave ----
__device__ inline void stage_chunk_async(const _Float16* __restrict__ K,
                                         const _Float16* __restrict__ V, int t0,
                                         _Float16 (*ldsK)[64], _Float16 (*ldsV)[32], int tid) {
  for (int c = tid; c < 256; c += 128) {          // 2 instructions / wave
    int row = c >> 3, col = (c & 7) << 3;
    async_copy16(&K[(size_t)(t0 + row) * D_K + col], &ldsK[row][col]);
  }
  for (int c = tid; c < 256; c += 128) {          // 2 instructions / wave
    int row = c >> 2, col = (c & 3) << 3;
    async_copy16(&V[(size_t)row * SEQ + t0 + col], &ldsV[row][col]);
  }
}

// ------------- Flash attention: 4 waves/block, 64 queries, double-buffered
// ------------- async global->LDS staging of shared K/V tiles -------------
__global__ __launch_bounds__(128) void attn_kernel(
    const _Float16* __restrict__ q, const _Float16* __restrict__ kbuf,
    const _Float16* __restrict__ vt, _Float16* __restrict__ zbuf) {
  __shared__ __align__(16) _Float16 ldsK[2][32][64];   // ping-pong K tiles
  __shared__ __align__(16) _Float16 ldsV[2][64][32];   // ping-pong V^T tiles
  __shared__ __align__(16) _Float16 pt[4][16][32];     // per-wave P tile (C->A transpose)
  int tid = threadIdx.x;
  int wave = tid >> 5, lane = tid & 31, lr = lane & 15, hf = lane >> 4;
  int q0 = blockIdx.x * 64 + wave * 16;
  int bh = blockIdx.y;
  int b_ = bh >> 4, h = bh & 15;
  const _Float16* Q = q    + (size_t)bh * SEQ * D_K;
  const _Float16* K = kbuf + (size_t)bh * SEQ * D_K;
  const _Float16* V = vt   + (size_t)bh * D_K * SEQ;
  v16h qa0 = load_a_rowmajor(Q, D_K, q0 + lr, 0, hf);
  v16h qa1 = load_a_rowmajor(Q, D_K, q0 + lr, 32, hf);
  float rm[8], rl[8];
  v8f zacc[4] = {};
#pragma unroll
  for (int r = 0; r < 8; r++) { rm[r] = -1e30f; rl[r] = 0.f; }

  stage_chunk_async(K, V, 0, ldsK[0], ldsV[0], tid);   // prologue: chunk 0 in flight

  for (int c = 0; c < CHUNKS; ++c) {
    int buf = c & 1;
    if (c + 1 < CHUNKS) {
      // issue next chunk into the other buffer, then wait for current chunk
      stage_chunk_async(K, V, (c + 1) * 32, ldsK[buf ^ 1], ldsV[buf ^ 1], tid);
      asm volatile("s_wait_asynccnt 4" ::: "memory");  // 4 newly issued remain in flight
    } else {
      asm volatile("s_wait_asynccnt 0" ::: "memory");
    }
    __syncthreads();                                   // all waves' current tiles visible

    // S = Q * K^T for 32 keys (two 16-key n-tiles, dk=64 = 2 chained WMMAs each)
    v8f s0 = {}, s1 = {};
    v16h b00 = load_b_kcontig(&ldsK[buf][0][0], D_K, lr,      hf * 16);
    v16h b01 = load_b_kcontig(&ldsK[buf][0][0], D_K, lr,      32 + hf * 16);
    v16h b10 = load_b_kcontig(&ldsK[buf][0][0], D_K, 16 + lr, hf * 16);
    v16h b11 = load_b_kcontig(&ldsK[buf][0][0], D_K, 16 + lr, 32 + hf * 16);
    s0 = __builtin_amdgcn_wmma_f32_16x16x32_f16(false, qa0, false, b00, (short)0, s0, false, false);
    s0 = __builtin_amdgcn_wmma_f32_16x16x32_f16(false, qa1, false, b01, (short)0, s0, false, false);
    s1 = __builtin_amdgcn_wmma_f32_16x16x32_f16(false, qa0, false, b10, (short)0, s1, false, false);
    s1 = __builtin_amdgcn_wmma_f32_16x16x32_f16(false, qa1, false, b11, (short)0, s1, false, false);

    // Online softmax; row M lives in VGPR r across one 16-lane half -> xor shuffles
#pragma unroll
    for (int r = 0; r < 8; r++) {
      float cmax = fmaxf(s0[r], s1[r]);
#pragma unroll
      for (int d = 1; d < 16; d <<= 1) cmax = fmaxf(cmax, __shfl_xor(cmax, d, 32));
      float mnew = fmaxf(rm[r], cmax);
      float alpha = __expf(rm[r] - mnew);
      rm[r] = mnew;
      float p0 = __expf(s0[r] - mnew);
      float p1 = __expf(s1[r] - mnew);
      float cs = p0 + p1;
#pragma unroll
      for (int d = 1; d < 16; d <<= 1) cs += __shfl_xor(cs, d, 32);
      rl[r] = rl[r] * alpha + cs;
#pragma unroll
      for (int j = 0; j < 4; j++) zacc[j][r] *= alpha;
      pt[wave][r + 8 * hf][lr]      = (_Float16)p0;   // C-layout scatter [m][key]
      pt[wave][r + 8 * hf][16 + lr] = (_Float16)p1;
    }
    // P tile is private to this wave; per-wave LDS ops are in-order -> a DS-count
    // drain + compiler fence is enough (no block barrier needed).
    asm volatile("s_wait_dscnt 0" ::: "memory");
    v16h pa;                                          // re-gather P in A layout
    ((uint4*)&pa)[0] = *(const uint4*)&pt[wave][lr][hf * 16];
    ((uint4*)&pa)[1] = *(const uint4*)&pt[wave][lr][hf * 16 + 8];
    // Z += P * V (K-dim = 32 keys) from the shared V^T tile
#pragma unroll
    for (int j = 0; j < 4; j++) {
      v16h vb = load_b_kcontig(&ldsV[buf][0][0], 32, j * 16 + lr, hf * 16);
      zacc[j] = __builtin_amdgcn_wmma_f32_16x16x32_f16(false, pa, false, vb, (short)0, zacc[j], false, false);
    }
    __syncthreads();   // all reads of buf done before iteration c+1 rewrites it
  }
  // normalize and write z in (b, t, h*64+dk) layout for the O-projection
#pragma unroll
  for (int j = 0; j < 4; j++) {
#pragma unroll
    for (int r = 0; r < 8; r++) {
      int trow = q0 + r + 8 * hf;
      float val = zacc[j][r] / rl[r];
      zbuf[((size_t)(b_ * SEQ + trow)) * D_MODEL + h * D_K + j * 16 + lr] = (_Float16)val;
    }
  }
}

// ------------- Output projection + bias + residual (fp32 out), 32x64 tile / wave -------------
__global__ __launch_bounds__(32) void oproj_kernel(
    const _Float16* __restrict__ zbuf, const _Float16* __restrict__ woT,
    const float* __restrict__ bo, const float* __restrict__ x, float* __restrict__ out) {
  int lane = threadIdx.x, lr = lane & 15, hf = lane >> 4;
  int m0 = blockIdx.y * 32, n0 = blockIdx.x * 64;
  v8f acc[2][4] = {};
  for (int kk = 0; kk < D_MODEL; kk += 32) {
    v16h a0 = load_a_rowmajor(zbuf, D_MODEL, m0 + lr,      kk, hf);
    v16h a1 = load_a_rowmajor(zbuf, D_MODEL, m0 + 16 + lr, kk, hf);
#pragma unroll
    for (int j = 0; j < 4; j++) {
      v16h b = load_b_kcontig(woT, D_MODEL, n0 + j * 16 + lr, kk + hf * 16);
      acc[0][j] = __builtin_amdgcn_wmma_f32_16x16x32_f16(false, a0, false, b, (short)0, acc[0][j], false, false);
      acc[1][j] = __builtin_amdgcn_wmma_f32_16x16x32_f16(false, a1, false, b, (short)0, acc[1][j], false, false);
    }
  }
#pragma unroll
  for (int j = 0; j < 4; j++) {
    int n = n0 + j * 16 + lr;
#pragma unroll
    for (int i = 0; i < 2; i++) {
#pragma unroll
      for (int r = 0; r < 8; r++) {
        int m = m0 + i * 16 + r + 8 * hf;
        out[(size_t)m * D_MODEL + n] = x[(size_t)m * D_MODEL + n] + acc[i][j][r] + bo[n];
      }
    }
  }
}

extern "C" void kernel_launch(void* const* d_in, const int* in_sizes, int n_in,
                              void* d_out, int out_size, void* d_ws, size_t ws_size,
                              hipStream_t stream) {
  const float* x  = (const float*)d_in[0];
  const float* wq = (const float*)d_in[1];
  const float* wk = (const float*)d_in[2];
  const float* wv = (const float*)d_in[3];
  const float* wo = (const float*)d_in[4];
  const float* bq = (const float*)d_in[5];
  const float* bk = (const float*)d_in[6];
  const float* bv = (const float*)d_in[7];
  const float* bo = (const float*)d_in[8];
  const float* g  = (const float*)d_in[9];
  const float* bt = (const float*)d_in[10];

  char* ws = (char*)d_ws;
  size_t off = 0;
  auto alloc = [&](size_t bytes) -> void* {
    void* p = ws + off;
    off += (bytes + 255) & ~(size_t)255;
    return p;
  };
  _Float16* xn  = (_Float16*)alloc((size_t)ROWS * D_MODEL * 2);                       // 8 MB
  _Float16* wT  = (_Float16*)alloc((size_t)NQKV * D_MODEL * 2);                       // 6 MB
  _Float16* woT = (_Float16*)alloc((size_t)D_MODEL * D_MODEL * 2);                    // 2 MB
  _Float16* qb  = (_Float16*)alloc((size_t)BATCH * NUM_HEADS * SEQ * D_K * 2);        // 8 MB
  _Float16* kbf = (_Float16*)alloc((size_t)BATCH * NUM_HEADS * SEQ * D_K * 2);        // 8 MB
  _Float16* vtb = (_Float16*)alloc((size_t)BATCH * NUM_HEADS * D_K * SEQ * 2);        // 8 MB
  _Float16* zb  = (_Float16*)alloc((size_t)ROWS * D_MODEL * 2);                       // 8 MB

  ln_kernel<<<ROWS, 256, 0, stream>>>(x, g, bt, xn);
  pack_wqkv<<<(NQKV * D_MODEL + 255) / 256, 256, 0, stream>>>(wq, wk, wv, wT);
  pack_wo<<<(D_MODEL * D_MODEL + 255) / 256, 256, 0, stream>>>(wo, woT);
  qkv_kernel<<<dim3(NQKV / 64, ROWS / 32), 32, 0, stream>>>(xn, wT, bq, bk, bv, qb, kbf, vtb);
  attn_kernel<<<dim3(SEQ / 64, BATCH * NUM_HEADS), 128, 0, stream>>>(qb, kbf, vtb, zb);
  oproj_kernel<<<dim3(D_MODEL / 64, ROWS / 32), 32, 0, stream>>>(zb, woT, bo, x, (float*)d_out);
}
